// PatchedSelfAttention_53592601919510
// MI455X (gfx1250) — matrix operand order repaired and checked by reading, not compiled, not run on hardware
//
#include <hip/hip_runtime.h>

// ---------------------------------------------------------------------------
// Fused patched self-attention block for MI455X (gfx1250, wave32, WMMA).
// One 256-thread workgroup per 4x4 patch; all GEMMs use
// v_wmma_f32_16x16x32_f16 with f32 accumulation; weights pre-converted to
// f16 column-major (B-operand-ready) in workspace.
// ---------------------------------------------------------------------------

typedef __attribute__((ext_vector_type(16))) _Float16 v16h;
typedef __attribute__((ext_vector_type(8)))  _Float16 v8h;
typedef __attribute__((ext_vector_type(8)))  float    v8f;

#define CH_    256
#define H_     128
#define W_     128
#define NPX_   32          // patches per row (128/4)
#define INK_   320         // padded IN_CH (295 -> 320, multiple of 32)
#define INCH_  295

// workspace (halfs) layout
#define WQ_OFF   0         // [256][320]
#define WKV_OFF  81920     // [512][320]
#define WMO_OFF  245760    // [256][256]
#define W1_OFF   311296    // [1024][256]
#define W2_OFF   573440    // [256][1024]
#define WS_HALFS 835584

#define OUT_ELEMS   (4 * 256 * 128 * 128)
#define COORD_ELEMS (4 * 3 * 128 * 128)

static __device__ __forceinline__ v8f wmma_f16f32(v16h a, v16h b, v8f c) {
  return __builtin_amdgcn_wmma_f32_16x16x32_f16(false, a, false, b, (short)0, c,
                                                false, false);
}

static __device__ __forceinline__ v8f vzero8() {
  v8f z;
#pragma unroll
  for (int i = 0; i < 8; ++i) z[i] = 0.0f;
  return z;
}

// A operand: 16xK row-major tile (stride in halfs).
// Layout: lane<16 -> row=lane, K kb+0..15;
//         lane>=16 -> row=lane-16, K kb+16..31.
static __device__ __forceinline__ v16h load_a16(const _Float16* base, int stride,
                                                int row, int kbase, int lane) {
  const int off = (lane < 16) ? 0 : 16;
  const _Float16* p = base + row * stride + kbase + off;
  return *(const v16h*)(p);
}

// B operand: KxN tile stored column-major (col*stride + k).
// Layout (contiguous k, as load_a16): lane<16 -> col=lane, K kb+0..15; lane>=16 -> col=lane-16, K kb+16..31.
static __device__ __forceinline__ v16h load_b16(const _Float16* base, int stride,
                                                int col, int kbase, int lane) {
  const _Float16* p = base + (long)col * stride + kbase + ((lane < 16) ? 0 : 16);
  return *(const v16h*)(p);
}

// ---------------------------------------------------------------------------
// Weight prep: f32 -> f16, transpose to column-major, zero-pad K to 320.
// q-input layout:  cols 0..38 = posenc, 39..294 = LN(inp)   (matches w_q rows)
// kv-input layout: cols 0..255 = inp,   256..294 = posenc   (matches w_kv rows)
// ---------------------------------------------------------------------------
__global__ void prep_weights_kernel(const float* __restrict__ wkv,
                                    const float* __restrict__ wq,
                                    const float* __restrict__ wmo,
                                    const float* __restrict__ w1,
                                    const float* __restrict__ w2,
                                    _Float16* __restrict__ ws) {
  for (int i = blockIdx.x * blockDim.x + threadIdx.x; i < WS_HALFS;
       i += gridDim.x * blockDim.x) {
    if (i < WKV_OFF) {                       // wq_t [256][320]
      int j = i, n = j / 320, k = j % 320;
      ws[WQ_OFF + j] = (k < INCH_) ? (_Float16)wq[k * 256 + n] : (_Float16)0.0f;
    } else if (i < WMO_OFF) {                // wkv_t [512][320]
      int j = i - WKV_OFF, n = j / 320, k = j % 320;
      ws[WKV_OFF + j] = (k < INCH_) ? (_Float16)wkv[k * 512 + n] : (_Float16)0.0f;
    } else if (i < W1_OFF) {                 // wmo_t [256][256]
      int j = i - WMO_OFF, n = j / 256, k = j % 256;
      ws[WMO_OFF + j] = (_Float16)wmo[k * 256 + n];
    } else if (i < W2_OFF) {                 // w1_t [1024][256]
      int j = i - W1_OFF, n = j / 256, k = j % 256;
      ws[W1_OFF + j] = (_Float16)w1[k * 1024 + n];
    } else {                                 // w2_t [256][1024]
      int j = i - W2_OFF, n = j / 1024, k = j % 1024;
      ws[W2_OFF + j] = (_Float16)w2[k * 256 + n];
    }
  }
}

// ---------------------------------------------------------------------------
// Main fused kernel: one workgroup per patch.
// ---------------------------------------------------------------------------
__global__ __launch_bounds__(256) void fused_patch_attn(
    const float* __restrict__ inp, const float* __restrict__ coords,
    const _Float16* __restrict__ ws,
    const float* __restrict__ b_mo, const float* __restrict__ b1,
    const float* __restrict__ b2, const float* __restrict__ g1,
    const float* __restrict__ beta1, const float* __restrict__ g2,
    const float* __restrict__ beta2, float* __restrict__ out) {
  // ---- LDS carve (lifetimes overlap-safe) --------------------------------
  __shared__ __align__(64) unsigned char smem[159744];
  _Float16* qin    = (_Float16*)(smem + 0);        // [16][320]  phase A..2
  _Float16* kvin   = (_Float16*)(smem + 10240);    // [48][320]  phase A..2
  _Float16* m_lds  = (_Float16*)(smem + 0);        // [16][256]  LN2..MLP1 (reuse)
  _Float16* h1_lds = (_Float16*)(smem + 8192);     // [16][1024] MLP1..MLP2 (reuse)
  _Float16* q_lds  = (_Float16*)(smem + 40960);    // [16][256]
  _Float16* k_lds  = (_Float16*)(smem + 49152);    // [48][256]  [key][d]
  _Float16* v_t    = (_Float16*)(smem + 73728);    // [256][64]  [d][key]
  float*    van    = (float*)(smem + 106496);      // [16][256]  raw patch (residual)
  _Float16* attn   = (_Float16*)(smem + 122880);   // [4][16][64]
  _Float16* o_attn = (_Float16*)(smem + 131072);   // [16][256]
  float*    o_lds  = (float*)(smem + 139264);      // [16][256]
  float*    scr    = (float*)(smem + 155648);      // scratch (<=1024 f32)
  float* s_rsum = scr;            // [16][16]
  float* s_rsq  = scr + 256;      // [16][16]
  float* s_mu   = scr + 512;      // [16]
  float* s_rs   = scr + 528;      // [16]
  float* s_cq   = scr + 544;      // [16][3]
  float* s_ckv  = scr + 592;      // [36][3]
  float* s_mean = scr + 700;      // [6] (q mean d, kv mean d)

  const _Float16* wq_t  = ws + WQ_OFF;
  const _Float16* wkv_t = ws + WKV_OFF;
  const _Float16* wmo_t = ws + WMO_OFF;
  const _Float16* w1_t  = ws + W1_OFF;
  const _Float16* w2_t  = ws + W2_OFF;

  const int p  = blockIdx.x;
  const int bb = p >> 10;            // batch
  const int pi = p & 1023;
  const int py = pi >> 5, px = pi & 31;
  const int t    = threadIdx.x;
  const int lane = t & 31;
  const int wave = t >> 5;
  const int colL = lane & 15;
  const int rbase = (lane < 16) ? 0 : 8;

  // ===================== Phase A: build inputs ============================
  const float* cbase = coords + (size_t)bb * 3 * H_ * W_;
  if (t < 108) {  // halo coords (zero-padded out of image)
    int r = t / 3, d = t % 3, i = r / 6, j = r % 6;
    int y = py * 4 - 1 + i, x = px * 4 - 1 + j;
    float v = (y >= 0 && y < H_ && x >= 0 && x < W_)
                  ? cbase[(size_t)d * H_ * W_ + y * W_ + x] : 0.0f;
    s_ckv[t] = v;
  }
  if (t < 48) {   // interior coords
    int r = t / 3, d = t % 3;
    int y = py * 4 + (r >> 2), x = px * 4 + (r & 3);
    s_cq[t] = cbase[(size_t)d * H_ * W_ + y * W_ + x];
  }
  __syncthreads();
  if (t < 6) {
    if (t < 3) {
      float s = 0.0f;
      for (int r = 0; r < 16; ++r) s += s_cq[r * 3 + t];
      s_mean[t] = s * (1.0f / 16.0f);
    } else {
      int d = t - 3; float s = 0.0f;
      for (int r = 0; r < 36; ++r) s += s_ckv[r * 3 + d];
      s_mean[t] = s * (1.0f / 36.0f);
    }
  }
  __syncthreads();
  if (t < 16) {   // q posenc -> qin cols 0..38, zero pad 295..319
    _Float16* row = qin + t * INK_;
#pragma unroll
    for (int d = 0; d < 3; ++d) {
      float c = s_cq[t * 3 + d] - s_mean[d];
      row[d] = (_Float16)c;
      float f = 1.0f;
      for (int k = 0; k < 6; ++k) {
        float xf = c * f;
        row[3 + d * 12 + k]     = (_Float16)__sinf(xf);
        row[3 + d * 12 + 6 + k] = (_Float16)__cosf(xf);
        f *= 2.0f;
      }
    }
    for (int c = INCH_; c < INK_; ++c) row[c] = (_Float16)0.0f;
  }
  if (t < 48) {   // kv posenc cols 256..294; zero pad cols + pad rows 36..47
    _Float16* row = kvin + t * INK_;
    if (t < 36) {
#pragma unroll
      for (int d = 0; d < 3; ++d) {
        float c = s_ckv[t * 3 + d] - s_mean[3 + d];
        row[256 + d] = (_Float16)c;
        float f = 1.0f;
        for (int k = 0; k < 6; ++k) {
          float xf = c * f;
          row[256 + 3 + d * 12 + k]     = (_Float16)__sinf(xf);
          row[256 + 3 + d * 12 + 6 + k] = (_Float16)__cosf(xf);
          f *= 2.0f;
        }
      }
      for (int c = INCH_; c < INK_; ++c) row[c] = (_Float16)0.0f;
    } else {
      for (int c = 0; c < INK_; ++c) row[c] = (_Float16)0.0f;
    }
  }
  { // raw input gather: thread t owns channel c=t
    const int c = t;
    const float* ic = inp + ((size_t)bb * CH_ + c) * H_ * W_;
#pragma unroll
    for (int ry = 0; ry < 4; ++ry) {
      float4 v4 = *(const float4*)(ic + (py * 4 + ry) * W_ + px * 4);
      van[(ry * 4 + 0) * 256 + c] = v4.x;
      van[(ry * 4 + 1) * 256 + c] = v4.y;
      van[(ry * 4 + 2) * 256 + c] = v4.z;
      van[(ry * 4 + 3) * 256 + c] = v4.w;
    }
    for (int i = 0; i < 6; ++i) {
      int y = py * 4 - 1 + i;
      bool yok = (y >= 0) && (y < H_);
      for (int j = 0; j < 6; ++j) {
        int x = px * 4 - 1 + j;
        float v = (yok && x >= 0 && x < W_) ? ic[y * W_ + x] : 0.0f;
        kvin[(i * 6 + j) * INK_ + c] = (_Float16)v;
      }
    }
    // zero v_t key-padding 48..63 (WMMA must never touch NaN garbage)
    for (int i = 0; i < 16; ++i) v_t[t * 64 + 48 + i] = (_Float16)0.0f;
  }
  __syncthreads();
  // LayerNorm-1 stats over channels, per row
  {
    int r = t >> 4, g = t & 15;
    float s = 0.0f, sq = 0.0f;
    for (int i = 0; i < 16; ++i) {
      float v = van[r * 256 + g * 16 + i];
      s += v; sq += v * v;
    }
    s_rsum[r * 16 + g] = s; s_rsq[r * 16 + g] = sq;
  }
  __syncthreads();
  if (t < 16) {
    float s = 0.0f, sq = 0.0f;
    for (int g = 0; g < 16; ++g) { s += s_rsum[t * 16 + g]; sq += s_rsq[t * 16 + g]; }
    float mu = s * (1.0f / 256.0f);
    float var = sq * (1.0f / 256.0f) - mu * mu;
    s_mu[t] = mu; s_rs[t] = rsqrtf(var + 1e-5f);
  }
  __syncthreads();
  {
    const int c = t;
    float gg = g1[c], bt = beta1[c];
    for (int r = 0; r < 16; ++r) {
      float v = (van[r * 256 + c] - s_mu[r]) * s_rs[r] * gg + bt;
      qin[r * INK_ + 39 + c] = (_Float16)v;
    }
  }
  __syncthreads();

  // ===================== Phase 2: q and kv projections ====================
  // 112 tile jobs: 0..15 q (16x16 tiles over N=256), 16..111 kv (3 M x 32 N)
  for (int job = wave; job < 112; job += 8) {
    v8f acc = vzero8();
    if (job < 16) {
      const int nt = job;
      for (int kc = 0; kc < 10; ++kc) {
        const int kb = kc * 32;
        v16h a = load_a16(qin, INK_, colL, kb, lane);
        v16h b = load_b16(wq_t, INK_, nt * 16 + colL, kb, lane);
        acc = wmma_f16f32(a, b, acc);
      }
      const int c = nt * 16 + colL;
#pragma unroll
      for (int j = 0; j < 8; ++j) q_lds[(rbase + j) * 256 + c] = (_Float16)acc[j];
    } else {
      const int jj = job - 16;
      const int nt = jj & 31, mt = jj >> 5;
      for (int kc = 0; kc < 10; ++kc) {
        const int kb = kc * 32;
        v16h a = load_a16(kvin, INK_, mt * 16 + colL, kb, lane);
        v16h b = load_b16(wkv_t, INK_, nt * 16 + colL, kb, lane);
        acc = wmma_f16f32(a, b, acc);
      }
      const int n = nt * 16 + colL;
      if (n < 256) {  // k: [key][d]
#pragma unroll
        for (int j = 0; j < 8; ++j)
          k_lds[(mt * 16 + rbase + j) * 256 + n] = (_Float16)acc[j];
      } else {        // v transposed: [d][key] (contiguous over rows)
        const int d = n - 256;
#pragma unroll
        for (int j = 0; j < 8; ++j)
          v_t[d * 64 + mt * 16 + rbase + j] = (_Float16)acc[j];
      }
    }
  }
  __syncthreads();

  // ===================== Phase 3: attention (waves 0..3 = heads) ==========
  if (wave < 4) {
    const int h = wave;
    v8f s0 = vzero8(), s1 = vzero8(), s2 = vzero8();
    for (int kc = 0; kc < 2; ++kc) {
      const int kb = h * 64 + kc * 32;
      v16h a = load_a16(q_lds, 256, colL, kb, lane);
      s0 = wmma_f16f32(a, load_b16(k_lds, 256,  0 + colL, kb, lane), s0);
      s1 = wmma_f16f32(a, load_b16(k_lds, 256, 16 + colL, kb, lane), s1);
      s2 = wmma_f16f32(a, load_b16(k_lds, 256, 32 + colL, kb, lane), s2);
    }
    _Float16* ah = attn + h * (16 * 64);
#pragma unroll
    for (int j = 0; j < 8; ++j) {
      float x0 = s0[j] * 0.125f;
      float x1 = s1[j] * 0.125f;
      float x2 = (colL < 4) ? s2[j] * 0.125f : -1e30f;  // keys 36..47 masked
      float mx = fmaxf(fmaxf(x0, x1), x2);
      for (int off = 8; off >= 1; off >>= 1)  // stays inside 16-lane halves
        mx = fmaxf(mx, __shfl_xor(mx, off, 32));
      float e0 = __expf(x0 - mx);
      float e1 = __expf(x1 - mx);
      float e2 = (colL < 4) ? __expf(x2 - mx) : 0.0f;
      float sm = e0 + e1 + e2;
      for (int off = 8; off >= 1; off >>= 1)
        sm += __shfl_xor(sm, off, 32);
      float inv = 1.0f / sm;
      const int r = rbase + j;
      ah[r * 64 +  0 + colL] = (_Float16)(e0 * inv);
      ah[r * 64 + 16 + colL] = (_Float16)(e1 * inv);
      ah[r * 64 + 32 + colL] = (_Float16)(e2 * inv);
      ah[r * 64 + 48 + colL] = (_Float16)0.0f;          // key padding 48..63
    }
    // intra-wave LDS store->load ordering before reusing ah as A operand
    __asm__ volatile("s_wait_dscnt 0" ::: "memory");
    for (int dt = 0; dt < 4; ++dt) {
      v8f acc = vzero8();
      for (int kc = 0; kc < 2; ++kc) {
        const int kb = kc * 32;
        v16h a = load_a16(ah, 64, colL, kb, lane);
        v16h b = load_b16(v_t, 64, h * 64 + dt * 16 + colL, kb, lane);
        acc = wmma_f16f32(a, b, acc);
      }
      const int d = h * 64 + dt * 16 + colL;
#pragma unroll
      for (int j = 0; j < 8; ++j)
        o_attn[(rbase + j) * 256 + d] = (_Float16)acc[j];
    }
  }
  __syncthreads();

  // ===================== Phase 4: w_mo + bias + residual + LN2 ============
  for (int nt = wave; nt < 16; nt += 8) {
    v8f acc = vzero8();
    for (int kc = 0; kc < 8; ++kc) {
      const int kb = kc * 32;
      v16h a = load_a16(o_attn, 256, colL, kb, lane);
      v16h b = load_b16(wmo_t, 256, nt * 16 + colL, kb, lane);
      acc = wmma_f16f32(a, b, acc);
    }
    const int c = nt * 16 + colL;
    const float bm = b_mo[c];
#pragma unroll
    for (int j = 0; j < 8; ++j) {
      const int r = rbase + j;
      o_lds[r * 256 + c] = acc[j] + bm + van[r * 256 + c];
    }
  }
  __syncthreads();
  {
    int r = t >> 4, g = t & 15;
    float s = 0.0f, sq = 0.0f;
    for (int i = 0; i < 16; ++i) {
      float v = o_lds[r * 256 + g * 16 + i];
      s += v; sq += v * v;
    }
    s_rsum[r * 16 + g] = s; s_rsq[r * 16 + g] = sq;
  }
  __syncthreads();
  if (t < 16) {
    float s = 0.0f, sq = 0.0f;
    for (int g = 0; g < 16; ++g) { s += s_rsum[t * 16 + g]; sq += s_rsq[t * 16 + g]; }
    float mu = s * (1.0f / 256.0f);
    float var = sq * (1.0f / 256.0f) - mu * mu;
    s_mu[t] = mu; s_rs[t] = rsqrtf(var + 1e-5f);
  }
  __syncthreads();
  {
    const int c = t;
    float gg = g2[c], bt = beta2[c];
    for (int r = 0; r < 16; ++r)
      m_lds[r * 256 + c] =
          (_Float16)((o_lds[r * 256 + c] - s_mu[r]) * s_rs[r] * gg + bt);
  }
  __syncthreads();

  // ===================== Phase 5: MLP up (256->1024) + GELU ===============
  for (int nt = wave; nt < 64; nt += 8) {
    v8f acc = vzero8();
    for (int kc = 0; kc < 8; ++kc) {
      const int kb = kc * 32;
      v16h a = load_a16(m_lds, 256, colL, kb, lane);
      v16h b = load_b16(w1_t, 256, nt * 16 + colL, kb, lane);
      acc = wmma_f16f32(a, b, acc);
    }
    const int n = nt * 16 + colL;
    const float bias = b1[n];
#pragma unroll
    for (int j = 0; j < 8; ++j) {
      float x = acc[j] + bias;
      float gel = 0.5f * x * (1.0f + erff(x * 0.70710678118f));  // exact gelu
      h1_lds[(rbase + j) * 1024 + n] = (_Float16)gel;
    }
  }
  __syncthreads();

  // ===================== Phase 6: MLP down (1024->256) + residual =========
  for (int nt = wave; nt < 16; nt += 8) {
    v8f acc = vzero8();
    for (int kc = 0; kc < 32; ++kc) {
      const int kb = kc * 32;
      v16h a = load_a16(h1_lds, 1024, colL, kb, lane);
      v16h b = load_b16(w2_t, 1024, nt * 16 + colL, kb, lane);
      acc = wmma_f16f32(a, b, acc);
    }
    const int c = nt * 16 + colL;
    const float bias = b2[c];
#pragma unroll
    for (int j = 0; j < 8; ++j) {
      const int r = rbase + j;
      o_lds[r * 256 + c] = acc[j] + bias + o_lds[r * 256 + c];
    }
  }
  __syncthreads();

  // ===================== Phase 7: fold back to NCHW =======================
  {
    const int c = t;
    float* ob = out + ((size_t)bb * CH_ + c) * H_ * W_;
#pragma unroll
    for (int ry = 0; ry < 4; ++ry) {
      float4 v4;
      v4.x = o_lds[(ry * 4 + 0) * 256 + c];
      v4.y = o_lds[(ry * 4 + 1) * 256 + c];
      v4.z = o_lds[(ry * 4 + 2) * 256 + c];
      v4.w = o_lds[(ry * 4 + 3) * 256 + c];
      *(float4*)(ob + (py * 4 + ry) * W_ + px * 4) = v4;
    }
  }
}

// ---------------------------------------------------------------------------
extern "C" void kernel_launch(void* const* d_in, const int* in_sizes, int n_in,
                              void* d_out, int out_size, void* d_ws,
                              size_t ws_size, hipStream_t stream) {
  (void)in_sizes; (void)n_in; (void)out_size; (void)ws_size;
  const float* inp    = (const float*)d_in[0];
  const float* coords = (const float*)d_in[1];
  const float* w_kv   = (const float*)d_in[2];
  const float* w_q    = (const float*)d_in[3];
  const float* w_mo   = (const float*)d_in[4];
  const float* b_mo   = (const float*)d_in[5];
  const float* w1     = (const float*)d_in[6];
  const float* b1     = (const float*)d_in[7];
  const float* w2     = (const float*)d_in[8];
  const float* b2     = (const float*)d_in[9];
  const float* g1     = (const float*)d_in[10];
  const float* beta1  = (const float*)d_in[11];
  const float* g2     = (const float*)d_in[12];
  const float* beta2  = (const float*)d_in[13];
  float* out = (float*)d_out;
  _Float16* ws = (_Float16*)d_ws;

  prep_weights_kernel<<<512, 256, 0, stream>>>(w_kv, w_q, w_mo, w1, w2, ws);
  fused_patch_attn<<<4096, 256, 0, stream>>>(inp, coords, ws, b_mo, b1, b2, g1,
                                             beta1, g2, beta2, out);
  // second tuple output: coords passthrough
  hipMemcpyAsync(out + OUT_ELEMS, coords, (size_t)COORD_ELEMS * sizeof(float),
                 hipMemcpyDeviceToDevice, stream);
}
